// VQCodebook_24429773980319
// MI455X (gfx1250) — compile-verified, matrix operand order checked by reading, not run on hardware
//
#include <hip/hip_runtime.h>
#include <hip/hip_bf16.h>

typedef __attribute__((ext_vector_type(16))) _Float16 v16h;
typedef __attribute__((ext_vector_type(8)))  _Float16 v8h;
typedef __attribute__((ext_vector_type(4)))  _Float16 v4h;
typedef __attribute__((ext_vector_type(8)))  float    v8f;

#define K_SLOTS 512
#define DIM 64
#define ROWS_PER_BLK 64
#define N_ROWS 65536
#define HW 4096
#define Z_Q_ELEMS 4194304   // 16*64*64*64
#define IDX_ELEMS 65536
#define LOG_K 6.2383246250395075f
#define SMEM_BYTES 273408

// Build a WMMA 16-bit A/B fragment for lane `L` from a row-contiguous LDS row.
// Per ISA 7.12.2: lane holds M/N = L&15; halves i<8 cover K = 8*(L>>4)+i,
// halves i>=8 cover K = 16 + 8*(L>>4) + (i-8).  hi = 8*(L>>4).
__device__ __forceinline__ v16h frag16(const _Float16* p, int hi) {
  v8h lo = *(const v8h*)(p + hi);
  v8h hv = *(const v8h*)(p + hi + 16);
  return __builtin_shufflevector(lo, hv, 0,1,2,3,4,5,6,7,8,9,10,11,12,13,14,15);
}

__global__ __launch_bounds__(128) void vq_fused_kernel(
    const float* __restrict__ ze, const float* __restrict__ cb,
    const float* __restrict__ u, float* __restrict__ out) {
  extern __shared__ __align__(16) char smem[];
  _Float16* cbRM = (_Float16*)smem;                    // [512][64]  64KB (reused as soft)
  _Float16* cbT  = cbRM + K_SLOTS * DIM;               // [64][512]  64KB
  float*    dist = (float*)(cbT + DIM * K_SLOTS);      // [64][512] 128KB (reused as zq stage)
  _Float16* zF   = (_Float16*)(dist + ROWS_PER_BLK * K_SLOTS); // [64][64] 8KB
  float*    cbN  = (float*)(zF + ROWS_PER_BLK * DIM);  // [512]
  float*    zN   = cbN + K_SLOTS;                      // [64]
  float*    zNp  = zN + ROWS_PER_BLK;                  // [2][64]
  float*    rcpSy= zNp + 2 * ROWS_PER_BLK;             // [64]
  _Float16* soft = cbRM;                               // alias (after GEMM1)
  float*    zqS  = dist;                               // alias (after pass2)

  const int t    = threadIdx.x;
  const int lane = t & 31;
  const int w    = t >> 5;                 // wave 0..3
  const int n0   = blockIdx.x * ROWS_PER_BLK;
  const int b    = n0 >> 12;               // image index
  const int hw0  = n0 & (HW - 1);
  const int hi   = (lane >> 4) * 8;        // half-wave K offset (in halves)
  const int l15  = lane & 15;
  const int rl   = w * 16 + l15;           // this wave's row for N-side / row-major ops

  // ---- codebook: load f32 (L2-resident), norms, convert to RM + T f16 ----
  for (int k = t; k < K_SLOTS; k += 128) {
    const float4* row = (const float4*)(cb + k * DIM);
    float nrm = 0.f;
#pragma unroll 4
    for (int j = 0; j < 16; ++j) {
      float4 v = row[j];
      float a4[4] = {v.x, v.y, v.z, v.w};
      int d0 = j * 4;
#pragma unroll
      for (int c = 0; c < 4; ++c) {
        nrm += a4[c] * a4[c];
        cbRM[k * DIM + d0 + c]      = (_Float16)a4[c];
        cbT[(d0 + c) * K_SLOTS + k] = (_Float16)a4[c];
      }
    }
    cbN[k] = nrm;
  }

  // ---- z tile: gather strided [bs,d,h,w] -> zF[row][dim] f16, row norms ----
  {
    int r = t & 63, dsel = t >> 6;
    float nrm = 0.f;
#pragma unroll 8
    for (int j = 0; j < 32; ++j) {
      int d = dsel + 2 * j;
      float v = ze[(((size_t)(b * DIM + d)) << 12) + hw0 + r];
      nrm += v * v;
      zF[r * DIM + d] = (_Float16)v;
    }
    zNp[dsel * ROWS_PER_BLK + r] = nrm;
  }
  __syncthreads();
  if (t < ROWS_PER_BLK) zN[t] = zNp[t] + zNp[ROWS_PER_BLK + t];
  __syncthreads();

  // ---- GEMM1: dist[row][code] = ||c||^2 + ||z||^2 - 2 * (cb @ z^T) ----
  // A = codebook tile (M=code, K=dim), B = z rows (K=dim, N=row)
  v16h bz0 = frag16(zF + rl * DIM + 0,  hi);
  v16h bz1 = frag16(zF + rl * DIM + 32, hi);
  float znr = zN[rl];
  for (int tt = 0; tt < 32; ++tt) {
    v8f acc = {};
    const _Float16* arow = cbRM + (tt * 16 + l15) * DIM;
    v16h a0 = frag16(arow, hi);
    v16h a1 = frag16(arow + 32, hi);
    acc = __builtin_amdgcn_wmma_f32_16x16x32_f16(false, a0, false, bz0, (short)0, acc, false, false);
    acc = __builtin_amdgcn_wmma_f32_16x16x32_f16(false, a1, false, bz1, (short)0, acc, false, false);
    // D: VGPR r -> M(code) = tt*16 + r + hi ; N(row) = l15 -> rl
#pragma unroll
    for (int r = 0; r < 8; ++r) {
      int code = tt * 16 + r + hi;
      dist[rl * K_SLOTS + code] = cbN[code] + znr - 2.0f * acc[r];
    }
  }
  __syncthreads();   // all waves done reading cbRM (soft aliases it)

  // ---- pass1: row max of y=(logit+gumbel)/T, argmax, max logit ----
  const int g = lane >> 4;                 // column half: g=0 -> k<256
  const size_t rowG = (size_t)(n0 + rl);
  const float4* uRow = (const float4*)(u + rowG * K_SLOTS);
  float bestY = -3.4e38f, mL = -3.4e38f;
  int bestK = 0;
  for (int j = 0; j < 64; ++j) {
    if (j < 56) __builtin_prefetch(uRow + g * 64 + j + 8, 0, 0);
    float4 uv = uRow[g * 64 + j];
    float4 dv = *(const float4*)(dist + rl * K_SLOTS + g * 256 + 4 * j);
    float ua[4] = {uv.x, uv.y, uv.z, uv.w};
    float da[4] = {dv.x, dv.y, dv.z, dv.w};
#pragma unroll
    for (int c = 0; c < 4; ++c) {
      float gm = -__logf(-__logf(ua[c] + 1e-10f) + 1e-10f);
      float l  = -da[c];
      float y  = (l + gm) * 2.0f;          // / TEMPERATURE(0.5)
      int   k  = g * 256 + 4 * j + c;
      if (y > bestY) { bestY = y; bestK = k; }
      mL = fmaxf(mL, l);
    }
  }
  { // pair reduce (lanes L and L^16 share a row)
    float oY = __shfl_xor(bestY, 16, 32);
    int   oK = __shfl_xor(bestK, 16, 32);
    if (oY > bestY || (oY == bestY && oK < bestK)) { bestY = oY; bestK = oK; }
    mL = fmaxf(mL, __shfl_xor(mL, 16, 32));
  }
  const float mY = bestY;

  // ---- pass2: exp/sums, store e_y as f16 (soft, pre-normalized), KL/commit ----
  float sY = 0.f, sL = 0.f, cAcc = 0.f, tAcc = 0.f;
  for (int j = 0; j < 64; ++j) {
    float4 uv = uRow[g * 64 + j];
    float4 dv = *(const float4*)(dist + rl * K_SLOTS + g * 256 + 4 * j);
    float ua[4] = {uv.x, uv.y, uv.z, uv.w};
    float da[4] = {dv.x, dv.y, dv.z, dv.w};
    v4h sv;
#pragma unroll
    for (int c = 0; c < 4; ++c) {
      float gm = -__logf(-__logf(ua[c] + 1e-10f) + 1e-10f);
      float l  = -da[c];
      float y  = (l + gm) * 2.0f;
      float ey = __expf(y - mY);
      sY += ey;
      float el = __expf(l - mL);
      sL += el;
      cAcc += el * da[c];
      tAcc += el * (l - mL);
      sv[c] = (_Float16)ey;
    }
    *(v4h*)(soft + rl * K_SLOTS + g * 256 + 4 * j) = sv;
  }
  sY   += __shfl_xor(sY,   16, 32);
  sL   += __shfl_xor(sL,   16, 32);
  cAcc += __shfl_xor(cAcc, 16, 32);
  tAcc += __shfl_xor(tAcc, 16, 32);
  float kl = 0.f, cm = 0.f;
  if (g == 0) {
    rcpSy[rl] = 1.0f / sY;
    out[Z_Q_ELEMS + rowG] = (float)bestK;        // hard index (as float)
    float lS = __logf(sL);
    kl = tAcc / sL - lS + LOG_K;                 // sum_k p*(log p + log K)
    cm = cAcc / sL;                              // sum_k p*dist
  }
#pragma unroll
  for (int off = 16; off > 0; off >>= 1) {
    kl += __shfl_xor(kl, off, 32);
    cm += __shfl_xor(cm, off, 32);
  }
  if (lane == 0) {
    atomicAdd(out + (Z_Q_ELEMS + IDX_ELEMS),     kl * (1.0f / 16.0f));
    atomicAdd(out + (Z_Q_ELEMS + IDX_ELEMS + 1), cm * (1.0f / 16.0f));
  }
  __syncthreads();   // all waves done reading dist (zqS aliases it)

  // ---- GEMM2: z_q[row][dim] = (e_y @ cb) * rcpSy[row] ----
  // A = soft rows (M=row, K=code), B = cbT (K=code, N=dim)
  v8f acc2[4] = {{}, {}, {}, {}};
  for (int ks = 0; ks < 16; ++ks) {
    v16h a = frag16(soft + rl * K_SLOTS + 32 * ks, hi);
#pragma unroll
    for (int nt = 0; nt < 4; ++nt) {
      v16h bb = frag16(cbT + (nt * 16 + l15) * K_SLOTS + 32 * ks, hi);
      acc2[nt] = __builtin_amdgcn_wmma_f32_16x16x32_f16(false, a, false, bb, (short)0, acc2[nt], false, false);
    }
  }
#pragma unroll
  for (int nt = 0; nt < 4; ++nt) {
#pragma unroll
    for (int r = 0; r < 8; ++r) {
      int row = w * 16 + r + hi;               // M = r + 8*(lane>>4)
      int d   = nt * 16 + l15;                 // N = lane&15
      zqS[row * DIM + d] = acc2[nt][r] * rcpSy[row];
    }
  }
  __syncthreads();

  // ---- coalesced z_q writeout: out[(b*64+d)*4096 + hw0 + row] ----
  {
    int r = t & 63, dsel = t >> 6;
#pragma unroll 8
    for (int j = 0; j < 32; ++j) {
      int d = dsel + 2 * j;
      out[(((size_t)(b * DIM + d)) << 12) + hw0 + r] = zqS[r * DIM + d];
    }
  }
}

extern "C" void kernel_launch(void* const* d_in, const int* in_sizes, int n_in,
                              void* d_out, int out_size, void* d_ws, size_t ws_size,
                              hipStream_t stream) {
  (void)in_sizes; (void)n_in; (void)d_ws; (void)ws_size; (void)out_size;
  const float* ze = (const float*)d_in[0];
  const float* cb = (const float*)d_in[1];
  const float* u  = (const float*)d_in[2];
  float* out = (float*)d_out;
  // zero the KL / commit_loss accumulator slots (graph-capture-safe async memset)
  hipMemsetAsync(out + (Z_Q_ELEMS + IDX_ELEMS), 0, 2 * sizeof(float), stream);
  dim3 grid(N_ROWS / ROWS_PER_BLK);   // 1024
  dim3 block(128);                    // 4 waves: one 16-row WMMA tile each
  hipLaunchKernelGGL(vq_fused_kernel, grid, block, SMEM_BYTES, stream, ze, cb, u, out);
}
